// GMTConvAttention_936302870759
// MI455X (gfx1250) — matrix operand order repaired and checked by reading, not compiled, not run on hardware
//
#include <hip/hip_runtime.h>
#include <hip/hip_bf16.h>

#define Bn    64
#define MAXNN 512
#define Dn    512
#define Hn    8
#define DHn   64
#define Sn    32
#define Nn    (Bn * MAXNN)   /* 32768 */
#define En    524288
#define NCH   16             /* feature chunks of 32 for LDS aggregation */

typedef __attribute__((ext_vector_type(16))) __bf16       bf16x16;
typedef __attribute__((ext_vector_type(8)))  float        f32x8;
typedef __attribute__((ext_vector_type(8)))  unsigned int u32x8;
typedef __attribute__((ext_vector_type(4)))  unsigned int u32x4;
typedef __attribute__((ext_vector_type(8)))  int          i32x8;
typedef __attribute__((ext_vector_type(4)))  int          i32x4;

static __device__ __forceinline__ f32x8 wmma_bf16(bf16x16 a, bf16x16 b, f32x8 c) {
  return __builtin_amdgcn_wmma_f32_16x16x32_bf16(false, a, false, b, (short)0, c, false, false);
}

// A-matrix (16x32, 16-bit) K offset for packed pair v, lane-half lh: lanes 0-15 K=0..15, lanes 16-31 K=16..31
static __device__ __forceinline__ int koffA(int v, int lh) {
  return lh * 16 + v * 2;
}
// B-matrix (32x16, 16-bit): lanes 0-15 hold K=0..15, lanes 16-31 K=16..31
static __device__ __forceinline__ int koffB(int v, int lh) {
  return lh * 16 + v * 2;
}

// pack two f32 into a bf16 pair (x -> low, y -> high) with one hardware cvt
static __device__ __forceinline__ unsigned pkbf(float x, float y) {
  unsigned r;
  asm("v_cvt_pk_bf16_f32 %0, %1, %2" : "=v"(r) : "v"(x), "v"(y));
  return r;
}

// A fragment from f32 source, features contiguous (global or LDS)
static __device__ __forceinline__ bf16x16 fragA_f32(const float* base, int lh) {
  u32x8 u;
#pragma unroll
  for (int v = 0; v < 8; ++v) {
    float2 p = *(const float2*)(base + koffA(v, lh));
    u[v] = pkbf(p.x, p.y);
  }
  return __builtin_bit_cast(bf16x16, u);
}

// B fragment from f32 source, K contiguous in memory (B^T rows: fixed n, features k)
static __device__ __forceinline__ bf16x16 fragB_f32(const float* base, int lh) {
  u32x8 u;
#pragma unroll
  for (int v = 0; v < 8; ++v) {
    float2 p = *(const float2*)(base + koffB(v, lh));
    u[v] = pkbf(p.x, p.y);
  }
  return __builtin_bit_cast(bf16x16, u);
}

// B fragment from f32 source with strided K (column-major tile)
static __device__ __forceinline__ bf16x16 fragB_f32_stride(const float* base, int stride, int lh) {
  u32x8 u;
#pragma unroll
  for (int v = 0; v < 8; ++v) {
    const int k0 = koffB(v, lh);
    u[v] = pkbf(base[(size_t)k0 * stride], base[(size_t)(k0 + 1) * stride]);
  }
  return __builtin_bit_cast(bf16x16, u);
}

// B fragment from transposed bf16 LDS tile: base = &Bt[col][0], K contiguous (2x ds_load_b128)
static __device__ __forceinline__ bf16x16 fragB_ldsT(const __bf16* base, int lh) {
  u32x8 u;
#pragma unroll
  for (int v = 0; v < 8; ++v)
    u[v] = *(const unsigned*)(base + lh * 16 + 2 * v);
  return __builtin_bit_cast(bf16x16, u);
}

static __device__ __forceinline__ float wave_max(float x) {
#pragma unroll
  for (int off = 16; off > 0; off >>= 1) x = fmaxf(x, __shfl_xor(x, off, 32));
  return x;
}
static __device__ __forceinline__ float wave_sum(float x) {
#pragma unroll
  for (int off = 16; off > 0; off >>= 1) x += __shfl_xor(x, off, 32);
  return x;
}

// ---------------- degree / norm ----------------
__global__ __launch_bounds__(256) void k_init_deg(float* __restrict__ deg) {
  int i = blockIdx.x * 256 + threadIdx.x;
  if (i < Nn) deg[i] = 1.0f;  // self-loop
}

__global__ __launch_bounds__(256) void k_edge_deg(const int* __restrict__ dst, float* __restrict__ deg) {
  int e = blockIdx.x * 256 + threadIdx.x;
  if (e < En) atomicAdd(&deg[dst[e]], 1.0f);
}

__global__ __launch_bounds__(256) void k_rsqrt(float* __restrict__ deg) {
  int i = blockIdx.x * 256 + threadIdx.x;
  if (i < Nn) deg[i] = rsqrtf(deg[i]);
}

// ---------------- edge bucketing by graph ----------------
__global__ __launch_bounds__(64) void k_zero_cnt(int* __restrict__ ecnt) {
  if (threadIdx.x < Bn) ecnt[threadIdx.x] = 0;
}

__global__ __launch_bounds__(256) void k_ecount(const int* __restrict__ src, int* __restrict__ ecnt) {
  int e = blockIdx.x * 256 + threadIdx.x;
  if (e < En) atomicAdd(&ecnt[src[e] >> 9], 1);
}

__global__ __launch_bounds__(64) void k_escan(const int* __restrict__ ecnt,
                                              int* __restrict__ eoff,
                                              int* __restrict__ ecur) {
  if (threadIdx.x == 0) {
    int run = 0;
    for (int i = 0; i < Bn; ++i) { eoff[i] = run; ecur[i] = run; run += ecnt[i]; }
    eoff[Bn] = run;
  }
}

__global__ __launch_bounds__(256) void k_ereorder(const int* __restrict__ src,
                                                  const int* __restrict__ dst,
                                                  const float* __restrict__ dinv,
                                                  int* __restrict__ ecur,
                                                  int* __restrict__ es_src,
                                                  int* __restrict__ es_dst,
                                                  float* __restrict__ es_nrm) {
  int e = blockIdx.x * 256 + threadIdx.x;
  if (e >= En) return;
  int s = src[e], d = dst[e];
  int g = s >> 9;                 // graph id (dst guaranteed in same graph)
  int pos = atomicAdd(&ecur[g], 1);
  es_src[pos] = s;                // global source node
  es_dst[pos] = d & 511;          // local destination node within graph
  es_nrm[pos] = dinv[s] * dinv[d];
}

// ---------------- fused GCN GEMMs: outK = X@Wk, outV = X@Wv (A tile via TDM) ----------------
__global__ __launch_bounds__(256) void k_gcn_gemm2(const float* __restrict__ X,
                                                   const float* __restrict__ Wk,
                                                   const float* __restrict__ Wv,
                                                   float* __restrict__ outK,
                                                   float* __restrict__ outV) {
  __shared__ float  As[128][32];     // 16 KB, filled by Tensor Data Mover
  __shared__ __bf16 Bkt[64][32];     // 4 KB, transposed weight tile
  __shared__ __bf16 Bvt[64][32];     // 4 KB
  const int tid = threadIdx.x;
  const int lane = tid & 31, w = tid >> 5;
  const int ll = lane & 15, lh = lane >> 4;
  const int row0 = (blockIdx.x >> 3) * 128;
  const int col0 = (blockIdx.x & 7) * 64;

  f32x8 accK[4] = {};
  f32x8 accV[4] = {};

  for (int kb = 0; kb < Dn; kb += 32) {
    if (w == 0) {
      // Tensor DMA descriptor (D#): 128x32 f32 tile of X at (row0, kb) -> As
      const unsigned lds_off = (unsigned)(size_t)(&As[0][0]);
      const unsigned long long ga =
          (unsigned long long)(size_t)(X + (size_t)row0 * Dn + kb);
      u32x4 g0;
      g0[0] = 1u;                                             // count=1, user descriptor
      g0[1] = lds_off;                                        // lds_addr
      g0[2] = (unsigned)(ga & 0xFFFFFFFFull);                 // global_addr[31:0]
      g0[3] = (unsigned)((ga >> 32) & 0x01FFFFFFull) | 0x80000000u;  // addr[56:32] | type=2
      i32x8 g1;
      g1[0] = 0x00020000;        // workgroup_mask=0, data_size=2 (4 bytes)
      g1[1] = (int)(512u << 16); // tensor_dim0 = 512 (elements per row)
      g1[2] = (int)0x80000000u;  // tensor_dim1 = 32768 (rows)
      g1[3] = (int)(32u << 16);  // tile_dim0 = 32
      g1[4] = 128;               // tile_dim1 = 128, tile_dim2 = 0
      g1[5] = 512;               // tensor_dim0_stride = 512
      g1[6] = 0;
      g1[7] = 0;
      i32x4 gz4 = {0, 0, 0, 0};
      i32x8 gz8 = {0, 0, 0, 0, 0, 0, 0, 0};
      __builtin_amdgcn_tensor_load_to_lds(g0, g1, gz4, gz4, gz8, 0);
    }
    // stage both weight tiles, transposed, bf16 (all 8 waves)
    for (int i = tid; i < 32 * 64; i += 256) {
      int r = i >> 6, c = i & 63;
      float wk = Wk[(size_t)(kb + r) * Dn + col0 + c];
      float wv = Wv[(size_t)(kb + r) * Dn + col0 + c];
      Bkt[c][r] = (__bf16)wk;
      Bvt[c][r] = (__bf16)wv;
    }
    if (w == 0) __builtin_amdgcn_s_wait_tensorcnt(0);
    __syncthreads();

    bf16x16 a = fragA_f32(&As[w * 16 + ll][0], lh);
    bf16x16 bk[4], bv[4];
#pragma unroll
    for (int t = 0; t < 4; ++t) {
      bk[t] = fragB_ldsT(&Bkt[t * 16 + ll][0], lh);
      bv[t] = fragB_ldsT(&Bvt[t * 16 + ll][0], lh);
    }
#pragma unroll
    for (int t = 0; t < 4; ++t) {
      accK[t] = wmma_bf16(a, bk[t], accK[t]);
      accV[t] = wmma_bf16(a, bv[t], accV[t]);
    }
    __syncthreads();
  }
#pragma unroll
  for (int t = 0; t < 4; ++t)
#pragma unroll
    for (int r = 0; r < 8; ++r) {
      size_t o = (size_t)(row0 + w * 16 + lh * 8 + r) * Dn + col0 + t * 16 + ll;
      outK[o] = accK[t][r];
      outV[o] = accV[t][r];
    }
}

// ---------------- GCN aggregation in LDS: one block per (graph, 32-feature chunk) --------
// out[g*512+n, f0+lane] = sum_edges(xw[src]*nrm) + xw[n]*dinv^2 + bias  (ds_add_f32 atomics)
__global__ __launch_bounds__(256) void k_gcn_aggr(const float* __restrict__ xw,
                                                  const float* __restrict__ dinv,
                                                  const float* __restrict__ bias,
                                                  const int* __restrict__ eoff,
                                                  const int* __restrict__ es_src,
                                                  const int* __restrict__ es_dst,
                                                  const float* __restrict__ es_nrm,
                                                  float* __restrict__ out) {
  __shared__ float acc[MAXNN * 32];  // 64 KB: 512 nodes x 32 features
  const int g = blockIdx.x >> 4;
  const int ch = blockIdx.x & (NCH - 1);
  const int tid = threadIdx.x, lane = tid & 31, w = tid >> 5;
  const int f0 = ch * 32;

  for (int i = tid; i < MAXNN * 32; i += 256) acc[i] = 0.f;
  __syncthreads();

  const int e0 = eoff[g], e1 = eoff[g + 1];
  for (int idx = e0 + w; idx < e1; idx += 8) {
    int sg = es_src[idx];
    int dl = es_dst[idx];
    float nr = es_nrm[idx];
    float val = xw[(size_t)sg * Dn + f0 + lane] * nr;
    atomicAdd(&acc[dl * 32 + lane], val);   // ds_add_f32
  }
  __syncthreads();

  for (int nn = w; nn < MAXNN; nn += 8) {
    int node = g * MAXNN + nn;
    float di = dinv[node];
    size_t o = (size_t)node * Dn + f0 + lane;
    out[o] = acc[nn * 32 + lane] + xw[o] * di * di + bias[f0 + lane];
  }
}

// ---------------- q projection (tiny): qs[s,e] = 0.125 * seed[s,:] . Wq[e,:] ----------------
__global__ __launch_bounds__(256) void k_qproj(const float* __restrict__ seed,
                                               const float* __restrict__ Wq,
                                               float* __restrict__ qs) {
  int idx = blockIdx.x * 256 + threadIdx.x;
  int s = idx >> 9, e = idx & 511;
  const float* sr = seed + (size_t)s * Dn;
  const float* wr = Wq + (size_t)e * Dn;
  float acc = 0.f;
  for (int d = 0; d < Dn; ++d) acc += sr[d] * wr[d];
  qs[idx] = acc * 0.125f;  // 1/sqrt(64)
}

// ---------------- attention: one block per (b, head) ----------------
__global__ __launch_bounds__(256) void k_attn(const float* __restrict__ qs,
                                              const float* __restrict__ kn,
                                              const float* __restrict__ vn,
                                              float* __restrict__ ctx) {
  __shared__ float sc[Sn][MAXNN];  // 64 KB
  const int b = blockIdx.x >> 3, hh = blockIdx.x & 7;
  const int tid = threadIdx.x, lane = tid & 31, w = tid >> 5;
  const int ll = lane & 15, lh = lane >> 4;
  const size_t nbase = (size_t)b * MAXNN;

  // scores = (q*scale) @ k^T  [32 x 512]; each wave: one s-strip, 8 n-tiles
  {
    const int stile = w >> 2;
    const int nt0 = (w & 3) * 8;
    f32x8 acc[8] = {};
#pragma unroll
    for (int kb = 0; kb < DHn; kb += 32) {
      bf16x16 a = fragA_f32(qs + (size_t)(stile * 16 + ll) * Dn + hh * DHn + kb, lh);
#pragma unroll
      for (int t = 0; t < 8; ++t) {
        const float* kbase = kn + (nbase + (nt0 + t) * 16 + ll) * Dn + hh * DHn + kb;
        acc[t] = wmma_bf16(a, fragB_f32(kbase, lh), acc[t]);
      }
    }
#pragma unroll
    for (int t = 0; t < 8; ++t)
#pragma unroll
      for (int r = 0; r < 8; ++r)
        sc[stile * 16 + lh * 8 + r][(nt0 + t) * 16 + ll] = acc[t][r];
  }
  __syncthreads();

  // softmax over 512 (4 rows per wave, wave32 shuffles)
#pragma unroll
  for (int rr = 0; rr < 4; ++rr) {
    int row = w * 4 + rr;
    float m = -3.0e38f;
    for (int j = lane; j < MAXNN; j += 32) m = fmaxf(m, sc[row][j]);
    m = wave_max(m);
    float ssum = 0.f;
    for (int j = lane; j < MAXNN; j += 32) {
      float ev = __expf(sc[row][j] - m);
      sc[row][j] = ev;
      ssum += ev;
    }
    ssum = wave_sum(ssum);
    float inv = 1.0f / ssum;
    for (int j = lane; j < MAXNN; j += 32) sc[row][j] *= inv;
  }
  __syncthreads();

  // ctx = attn @ v  [32 x 64], one 16x16 tile per wave
  {
    const int stile = w & 1, dtile = w >> 1;
    const int dcol = dtile * 16 + ll;
    f32x8 acc = {};
    for (int kb = 0; kb < MAXNN; kb += 32) {
      bf16x16 a = fragA_f32(&sc[stile * 16 + ll][kb], lh);
      bf16x16 bb = fragB_f32_stride(vn + (nbase + kb) * Dn + hh * DHn + dcol, Dn, lh);
      acc = wmma_bf16(a, bb, acc);
    }
#pragma unroll
    for (int r = 0; r < 8; ++r) {
      int s = stile * 16 + lh * 8 + r;
      ctx[((size_t)b * Sn + s) * Dn + hh * DHn + dcol] = acc[r];
    }
  }
}

// ---------------- fused output proj + LN + FF + LN, one block per graph ----------------
__global__ __launch_bounds__(256) void k_epilogue(const float* __restrict__ ctx,
                                                  const float* __restrict__ seed,
                                                  const float* __restrict__ Wo,
                                                  const float* __restrict__ bo,
                                                  const float* __restrict__ Wff,
                                                  const float* __restrict__ bff,
                                                  const float* __restrict__ gh,
                                                  const float* __restrict__ bh,
                                                  const float* __restrict__ gz,
                                                  const float* __restrict__ bz,
                                                  float* __restrict__ out) {
  __shared__ float hb[Sn][Dn];  // 64 KB
  const int b = blockIdx.x;
  const int tid = threadIdx.x, lane = tid & 31, w = tid >> 5;
  const int ll = lane & 15, lh = lane >> 4;
  const float* cb = ctx + (size_t)b * Sn * Dn;
  const int stile = w >> 2;
  const int nc0 = (w & 3) * 8;
  const int srow = stile * 16 + ll;

  // GEMM1: hb = ctx_b @ Wo^T + bo + seed
  {
    f32x8 acc[8] = {};
    for (int kb = 0; kb < Dn; kb += 32) {
      bf16x16 a = fragA_f32(cb + (size_t)srow * Dn + kb, lh);
#pragma unroll
      for (int t = 0; t < 8; ++t) {
        int ncol = (nc0 + t) * 16 + ll;
        acc[t] = wmma_bf16(a, fragB_f32(Wo + (size_t)ncol * Dn + kb, lh), acc[t]);
      }
    }
#pragma unroll
    for (int t = 0; t < 8; ++t) {
      int ncol = (nc0 + t) * 16 + ll;
#pragma unroll
      for (int r = 0; r < 8; ++r) {
        int s = stile * 16 + lh * 8 + r;
        hb[s][ncol] = acc[t][r] + bo[ncol] + seed[(size_t)s * Dn + ncol];
      }
    }
  }
  __syncthreads();

  // LN1
#pragma unroll
  for (int rr = 0; rr < 4; ++rr) {
    int row = w * 4 + rr;
    float sum = 0.f;
    for (int j = lane; j < Dn; j += 32) sum += hb[row][j];
    sum = wave_sum(sum);
    float mean = sum * (1.0f / Dn);
    float vs = 0.f;
    for (int j = lane; j < Dn; j += 32) { float dd = hb[row][j] - mean; vs += dd * dd; }
    vs = wave_sum(vs);
    float rs = rsqrtf(vs * (1.0f / Dn) + 1e-5f);
    for (int j = lane; j < Dn; j += 32)
      hb[row][j] = (hb[row][j] - mean) * rs * gh[j] + bh[j];
  }
  __syncthreads();

  // GEMM2 (accumulators in registers): ff = hb @ Wff^T
  f32x8 accs[8] = {};
  for (int kb = 0; kb < Dn; kb += 32) {
    bf16x16 a = fragA_f32(&hb[srow][kb], lh);
#pragma unroll
    for (int t = 0; t < 8; ++t) {
      int ncol = (nc0 + t) * 16 + ll;
      accs[t] = wmma_bf16(a, fragB_f32(Wff + (size_t)ncol * Dn + kb, lh), accs[t]);
    }
  }
  __syncthreads();
#pragma unroll
  for (int t = 0; t < 8; ++t) {
    int ncol = (nc0 + t) * 16 + ll;
#pragma unroll
    for (int r = 0; r < 8; ++r) {
      int s = stile * 16 + lh * 8 + r;
      hb[s][ncol] = hb[s][ncol] + accs[t][r] + bff[ncol];
    }
  }
  __syncthreads();

  // LN2 -> out
#pragma unroll
  for (int rr = 0; rr < 4; ++rr) {
    int row = w * 4 + rr;
    float sum = 0.f;
    for (int j = lane; j < Dn; j += 32) sum += hb[row][j];
    sum = wave_sum(sum);
    float mean = sum * (1.0f / Dn);
    float vs = 0.f;
    for (int j = lane; j < Dn; j += 32) { float dd = hb[row][j] - mean; vs += dd * dd; }
    vs = wave_sum(vs);
    float rs = rsqrtf(vs * (1.0f / Dn) + 1e-5f);
    float* orow = out + ((size_t)b * Sn + row) * Dn;
    for (int j = lane; j < Dn; j += 32)
      orow[j] = (hb[row][j] - mean) * rs * gz[j] + bz[j];
  }
}

extern "C" void kernel_launch(void* const* d_in, const int* in_sizes, int n_in,
                              void* d_out, int out_size, void* d_ws, size_t ws_size,
                              hipStream_t stream) {
  (void)in_sizes; (void)n_in; (void)out_size; (void)ws_size;
  const float* x    = (const float*)d_in[1];
  const int*   ei   = (const int*)d_in[2];
  const float* seed = (const float*)d_in[4];
  const float* Wk   = (const float*)d_in[5];
  const float* bk   = (const float*)d_in[6];
  const float* Wv   = (const float*)d_in[7];
  const float* bv   = (const float*)d_in[8];
  const float* Wq   = (const float*)d_in[9];
  const float* Wo   = (const float*)d_in[10];
  const float* bo   = (const float*)d_in[11];
  const float* Wff  = (const float*)d_in[12];
  const float* bff  = (const float*)d_in[13];
  const float* gh   = (const float*)d_in[14];
  const float* bh   = (const float*)d_in[15];
  const float* gz   = (const float*)d_in[16];
  const float* bz   = (const float*)d_in[17];
  float* out = (float*)d_out;

  const int* src = ei;
  const int* dst = ei + En;

  float* ws  = (float*)d_ws;
  float* xwk = ws;                        // N*D
  float* xwv = xwk + (size_t)Nn * Dn;     // N*D
  float* kn  = xwv + (size_t)Nn * Dn;     // N*D
  float* vn  = kn + (size_t)Nn * Dn;      // N*D
  float* deg = vn + (size_t)Nn * Dn;      // N  (becomes dinv)
  float* qs  = deg + Nn;                  // S*D
  float* ctx = qs + Sn * Dn;              // B*S*D
  int*   es_src = (int*)(ctx + (size_t)Bn * Sn * Dn);  // E
  int*   es_dst = es_src + En;                         // E
  float* es_nrm = (float*)(es_dst + En);               // E
  int*   ecnt   = (int*)(es_nrm + En);                 // B
  int*   eoff   = ecnt + Bn;                           // B+1
  int*   ecur   = eoff + Bn + 1;                       // B

  k_init_deg<<<Nn / 256, 256, 0, stream>>>(deg);
  k_edge_deg<<<En / 256, 256, 0, stream>>>(dst, deg);
  k_rsqrt<<<Nn / 256, 256, 0, stream>>>(deg);

  // bucket edges by graph for LDS-resident aggregation
  k_zero_cnt<<<1, 64, 0, stream>>>(ecnt);
  k_ecount<<<En / 256, 256, 0, stream>>>(src, ecnt);
  k_escan<<<1, 64, 0, stream>>>(ecnt, eoff, ecur);
  k_ereorder<<<En / 256, 256, 0, stream>>>(src, dst, deg, ecur, es_src, es_dst, es_nrm);

  const int gemm_grid = (Nn / 128) * (Dn / 64);  // 2048
  k_gcn_gemm2<<<gemm_grid, 256, 0, stream>>>(x, Wk, Wv, xwk, xwv);

  k_gcn_aggr<<<Bn * NCH, 256, 0, stream>>>(xwk, deg, bk, eoff, es_src, es_dst, es_nrm, kn);
  k_gcn_aggr<<<Bn * NCH, 256, 0, stream>>>(xwv, deg, bv, eoff, es_src, es_dst, es_nrm, vn);

  k_qproj<<<(Sn * Dn) / 256, 256, 0, stream>>>(seed, Wq, qs);
  k_attn<<<Bn * Hn, 256, 0, stream>>>(qs, kn, vn, ctx);
  k_epilogue<<<Bn, 256, 0, stream>>>(ctx, seed, Wo, bo, Wff, bff, gh, bh, gz, bz, out);
}
